// InvarLayer_77884936945789
// MI455X (gfx1250) — matrix operand order repaired and checked by reading, not compiled, not run on hardware
//
#include <hip/hip_runtime.h>
#include <hip/hip_bf16.h>

// ---------------- problem constants ----------------
// x:[16,64,112,112] f32; w1:[128,384]; b1:[128]; w2:[128,128]; b2:[128]
// out:[16,128,112,112] f32
#define NB    16
#define CIN   64
#define IH    112
#define IW    112
#define HWPIX 12544          // 112*112
#define COUT  128
#define KTOT  384            // 6*CIN
#define BSTR  40             // sBt row stride in halves (80B, mult of 16B)

typedef __attribute__((ext_vector_type(16))) _Float16 v16h;
typedef __attribute__((ext_vector_type(8)))  _Float16 v8h;
typedef __attribute__((ext_vector_type(4)))  _Float16 v4h;
typedef __attribute__((ext_vector_type(8)))  float    v8f;

union Frag16 { v16h v; v8h h[2]; };

// ---------------------------------------------------------------
// Kernel 0: zero the per-batch max-abs accumulators (48 uints)
// ---------------------------------------------------------------
__global__ void invar_init_m(unsigned* m) {
    if (threadIdx.x < 3 * NB) m[threadIdx.x] = 0u;
}

// ---------------------------------------------------------------
// Kernel 1/2: depthwise 7x7 Gaussian-derivative stencils.
// stage==0: in = x, writes ux,uy,uxx,uyy; max-abs into m0/m1/m2.
// stage==1: in = uy, writes uxy (= kdx stencil); max-abs into m2.
// One block = 16x16 output tile of one (b,c) plane, LDS halo tile.
// ---------------------------------------------------------------
__global__ __launch_bounds__(256) void invar_conv7(
        const float* __restrict__ in,
        float* __restrict__ o0, float* __restrict__ o1,
        float* __restrict__ o2, float* __restrict__ o3,
        unsigned* __restrict__ mbuf, int stage) {
    __shared__ float tile[22][24];
    __shared__ float st[4][49];
    __shared__ unsigned smax[3];

    const int tid   = threadIdx.x;
    const int plane = blockIdx.x / 49;           // b*CIN + c
    const int t     = blockIdx.x % 49;
    const int b     = plane >> 6;
    const int ty0   = (t / 7) * 16;
    const int tx0   = (t % 7) * 16;
    const float e2  = 0.99f * 0.99f;

    if (tid < 49) {
        const int i = tid / 7, j = tid % 7;
        const float X = (float)(j - 3);
        const float Y = (float)(3 - i);          // Y = -row, up positive
        const float g = __expf(-e2 * (X * X + Y * Y));
        st[0][tid] = -2.f * e2 * X * g;                         // kdx
        st[1][tid] = -2.f * e2 * Y * g;                         // kdy
        st[2][tid] = (4.f * e2 * e2 * X * X - 2.f * e2) * g;    // kdxx
        st[3][tid] = (4.f * e2 * e2 * Y * Y - 2.f * e2) * g;    // kdyy
    }
    if (tid < 3) smax[tid] = 0u;

    const float* ip = in + (size_t)plane * HWPIX;
    for (int idx = tid; idx < 22 * 22; idx += 256) {
        const int r = idx / 22, c = idx % 22;
        const int gy = ty0 + r - 3, gx = tx0 + c - 3;
        float v = 0.f;
        if (gy >= 0 && gy < IH && gx >= 0 && gx < IW) v = ip[gy * IW + gx];
        tile[r][c] = v;
    }
    __syncthreads();

    const int ty = tid >> 4, tx = tid & 15;
    float a0 = 0.f, a1 = 0.f, a2 = 0.f, a3 = 0.f;
#pragma unroll
    for (int i = 0; i < 7; ++i)
#pragma unroll
        for (int j = 0; j < 7; ++j) {
            const float v = tile[ty + i][tx + j];
            a0 += v * st[0][i * 7 + j];
            a1 += v * st[1][i * 7 + j];
            a2 += v * st[2][i * 7 + j];
            a3 += v * st[3][i * 7 + j];
        }

    const size_t o = (size_t)plane * HWPIX + (size_t)(ty0 + ty) * IW + (tx0 + tx);
    if (stage == 0) {
        o0[o] = a0; o1[o] = a1; o2[o] = a2; o3[o] = a3;
        const float c0 = fabsf(tile[ty + 3][tx + 3]);            // |x| at this pixel
        const float c1 = fmaxf(fabsf(a0), fabsf(a1));            // |ux|,|uy|
        const float c2 = fmaxf(fabsf(a2), fabsf(a3));            // |uxx|,|uyy|
        atomicMax(&smax[0], __float_as_uint(c0));
        atomicMax(&smax[1], __float_as_uint(c1));
        atomicMax(&smax[2], __float_as_uint(c2));
    } else {
        o0[o] = a0;                                              // uxy = kdx * uy
        atomicMax(&smax[2], __float_as_uint(fabsf(a0)));
    }
    __syncthreads();
    if (tid < 3 && smax[tid]) atomicMax(&mbuf[b * 3 + tid], smax[tid]);
}

// ---------------------------------------------------------------
// Kernel 3: convert weights to f16 (keep native [out][k] layout,
// which is exactly the K-innermost order the B-fragments want).
// ---------------------------------------------------------------
__global__ __launch_bounds__(256) void invar_pack_w(
        const float* __restrict__ w1, const float* __restrict__ w2,
        _Float16* __restrict__ w1h, _Float16* __restrict__ w2h) {
    const int idx = blockIdx.x * 256 + threadIdx.x;
    if (idx < COUT * KTOT) w1h[idx] = (_Float16)w1[idx];
    if (idx < COUT * COUT) w2h[idx] = (_Float16)w2[idx];
}

// ---------------------------------------------------------------
// Kernel 4: fused normalize -> GEMM1(+b1,ReLU) -> GEMM2(+b2)
// One block = one batch b, 64-pixel tile. 256 threads = 8 waves.
// Wave w: M-subtile (w&3)*16, N half (w>>2)*64 -> 4 WMMA C-tiles.
// Weight chunks (32 K) are double-buffered in LDS: one barrier per
// K-step, next chunk's global loads issued before current WMMAs.
// Fragment layouts per CDNA5 ISA 7.12.2 (wave32, 16-bit A/B, f32 C);
// A/B fragments are loaded as 2x ds_load_b128 each.
// ---------------------------------------------------------------
__global__ __launch_bounds__(256) void invar_gemm(
        const float* __restrict__ x,
        const float* __restrict__ ux,  const float* __restrict__ uy,
        const float* __restrict__ uxx, const float* __restrict__ uxy,
        const float* __restrict__ uyy,
        const _Float16* __restrict__ w1h, const float* __restrict__ b1,
        const _Float16* __restrict__ w2h, const float* __restrict__ b2,
        const unsigned* __restrict__ mbuf, float* __restrict__ out) {
    // sA: A tile [row][k], stride 392 halves (784B, mult of 16B);
    //     reused as h [row][136] f16, then as out-bounce [n][68] f32.
    // sBt: double-buffered weight chunk, K-innermost: [buf][n][BSTR].
    __shared__ __align__(16) _Float16 sA[64 * 392];
    __shared__ __align__(16) _Float16 sBt[2][128 * BSTR];

    const int tid   = threadIdx.x;
    const int b     = blockIdx.x / 196;
    const int tileI = blockIdx.x % 196;
    const int pix0  = tileI * 64;

    // per-batch normalization scales (fold into A so |A| <= 1 -> safe in f16)
    const float s0 = 1.0f / __uint_as_float(mbuf[b * 3 + 0]);
    const float s1 = 1.0f / __uint_as_float(mbuf[b * 3 + 1]);
    const float s2 = 1.0f / __uint_as_float(mbuf[b * 3 + 2]);
    const float* bufs[6] = { x, ux, uy, uxx, uxy, uyy };
    const float  scl[6]  = { s0, s1, s1, s2, s2, s2 };

    const int lane  = tid & 31;
    const int wave  = tid >> 5;
    const int rm    = (wave & 3) * 16;      // M subtile base row
    const int nb    = (wave >> 2) * 64;     // N base (0 or 64)
    const int lhalf = lane >> 4;            // 0: lanes 0-15, 1: lanes 16-31
    const int l15   = lane & 15;

    // copy-lane mapping for weight chunks: 8B per op, 4 ops per thread
    const int cpN  = tid >> 1;                       // 0..127
    const int cpK  = (tid & 1) * 16;                 // 0 or 16

    // hoist biases for this thread's 4 N-subtiles
    float bv1[4], bv2[4];
#pragma unroll
    for (int sub = 0; sub < 4; ++sub) {
        bv1[sub] = b1[nb + sub * 16 + l15];
        bv2[sub] = b2[nb + sub * 16 + l15];
    }

    // ---- build normalized invariant tile: A[64 pixels][384 k] in f16 ----
    for (int idx = tid; idx < KTOT * 16; idx += 256) {
        const int k  = idx >> 4;          // 0..383
        const int r4 = (idx & 15) * 4;    // row group 0,4,...,60
        const int g  = k / CIN;
        const int c  = k - g * CIN;
        const float4 v = *(const float4*)&bufs[g][((size_t)(b * CIN + c)) * HWPIX + pix0 + r4];
        const float s = scl[g];
        sA[(r4 + 0) * 392 + k] = (_Float16)(v.x * s);
        sA[(r4 + 1) * 392 + k] = (_Float16)(v.y * s);
        sA[(r4 + 2) * 392 + k] = (_Float16)(v.z * s);
        sA[(r4 + 3) * 392 + k] = (_Float16)(v.w * s);
    }

    // prologue: weight chunk 0 into buffer 0
#pragma unroll
    for (int q = 0; q < 4; ++q) {
        const int kk4 = cpK + q * 4;
        *(v4h*)&sBt[0][cpN * BSTR + kk4] = *(const v4h*)&w1h[cpN * KTOT + kk4];
    }

    // ================= GEMM1: [64x384] x [384x128] =================
    v8f acc[4] = {};
    for (int kt = 0; kt < 12; ++kt) {
        __syncthreads();   // chunk kt ready; buffer (kt+1)&1 free
        // issue next chunk's global loads before current compute
        if (kt < 11) {
            const int k0n = (kt + 1) * 32;
#pragma unroll
            for (int q = 0; q < 4; ++q) {
                const int kk4 = cpK + q * 4;
                *(v4h*)&sBt[(kt + 1) & 1][cpN * BSTR + kk4] =
                    *(const v4h*)&w1h[cpN * KTOT + k0n + kk4];
                if (kt < 10)
                    __builtin_prefetch(&w1h[cpN * KTOT + k0n + 32 + kk4], 0, 3);
            }
        }
        // A frag: lane row = rm+l15; K runs kb..kb+7 and kb+16..kb+23 (contiguous)
        const int kb = kt * 32 + lhalf * 8;
        const _Float16* ar = &sA[(rm + l15) * 392];
        Frag16 a;
        a.h[0] = *(const v8h*)(ar + kb);
        a.h[1] = *(const v8h*)(ar + kb + 16);

        const _Float16* bbuf = sBt[kt & 1];
#pragma unroll
        for (int sub = 0; sub < 4; ++sub) {
            const int n = nb + sub * 16 + l15;                   // B col = lane
            const _Float16* br = &bbuf[n * BSTR + lhalf * 8];    // K rows kb..kb+7, kb+16..kb+23
            Frag16 bf;
            bf.h[0] = *(const v8h*)(br);
            bf.h[1] = *(const v8h*)(br + 16);
            acc[sub] = __builtin_amdgcn_wmma_f32_16x16x32_f16(
                false, a.v, false, bf.v, (short)0, acc[sub], false, false);
        }
    }

    // ---- h = relu(acc + b1) -> back to LDS (sA reused) as f16 [64][136] ----
    __syncthreads();
#pragma unroll
    for (int sub = 0; sub < 4; ++sub) {
        const int n = nb + sub * 16 + l15;
#pragma unroll
        for (int r = 0; r < 8; ++r) {
            const int row = rm + r + lhalf * 8;                  // C layout: VGPR r, M = r (+8 hi lanes)
            sA[row * 136 + n] = (_Float16)fmaxf(acc[sub][r] + bv1[sub], 0.0f);
        }
    }

    // prologue for GEMM2 weights
#pragma unroll
    for (int q = 0; q < 4; ++q) {
        const int kk4 = cpK + q * 4;
        *(v4h*)&sBt[0][cpN * BSTR + kk4] = *(const v4h*)&w2h[cpN * COUT + kk4];
    }

    // ================= GEMM2: [64x128] x [128x128] =================
    v8f acc2[4] = {};
    for (int kt = 0; kt < 4; ++kt) {
        __syncthreads();
        if (kt < 3) {
            const int k0n = (kt + 1) * 32;
#pragma unroll
            for (int q = 0; q < 4; ++q) {
                const int kk4 = cpK + q * 4;
                *(v4h*)&sBt[(kt + 1) & 1][cpN * BSTR + kk4] =
                    *(const v4h*)&w2h[cpN * COUT + k0n + kk4];
            }
        }
        const int kb = kt * 32 + lhalf * 8;
        const _Float16* ar = &sA[(rm + l15) * 136];
        Frag16 a;
        a.h[0] = *(const v8h*)(ar + kb);
        a.h[1] = *(const v8h*)(ar + kb + 16);

        const _Float16* bbuf = sBt[kt & 1];
#pragma unroll
        for (int sub = 0; sub < 4; ++sub) {
            const int n = nb + sub * 16 + l15;
            const _Float16* br = &bbuf[n * BSTR + lhalf * 8];
            Frag16 bf;
            bf.h[0] = *(const v8h*)(br);
            bf.h[1] = *(const v8h*)(br + 16);
            acc2[sub] = __builtin_amdgcn_wmma_f32_16x16x32_f16(
                false, a.v, false, bf.v, (short)0, acc2[sub], false, false);
        }
    }

    // ---- bounce C through LDS to get coalesced stores: sO[n][row] f32 ----
    __syncthreads();                       // all h reads done; sA reusable
    float* sO = (float*)sA;                // [128][68] floats, 272B stride
#pragma unroll
    for (int sub = 0; sub < 4; ++sub) {
        const int n = nb + sub * 16 + l15;
#pragma unroll
        for (int r = 0; r < 8; ++r) {
            const int row = rm + r + lhalf * 8;
            sO[n * 68 + row] = acc2[sub][r] + bv2[sub];
        }
    }
    __syncthreads();
    // cooperative coalesced writeout: 64 consecutive pixels per channel
    for (int idx = tid; idx < COUT * 16; idx += 256) {
        const int n  = idx >> 4;
        const int r4 = (idx & 15) * 4;
        const float4 v = *(const float4*)&sO[n * 68 + r4];
        *(float4*)&out[((size_t)(b * COUT + n)) * HWPIX + pix0 + r4] = v;
    }
}

// ---------------------------------------------------------------
// Host launcher
// ---------------------------------------------------------------
extern "C" void kernel_launch(void* const* d_in, const int* in_sizes, int n_in,
                              void* d_out, int out_size, void* d_ws, size_t ws_size,
                              hipStream_t stream) {
    (void)in_sizes; (void)n_in; (void)out_size; (void)ws_size;

    const float* x  = (const float*)d_in[0];   // [16,64,112,112]
    const float* w1 = (const float*)d_in[1];   // [128,384]
    const float* b1 = (const float*)d_in[2];   // [128]
    const float* w2 = (const float*)d_in[3];   // [128,128]
    const float* b2 = (const float*)d_in[4];   // [128]
    float* out = (float*)d_out;                // [16,128,112,112]

    // workspace layout
    char* ws = (char*)d_ws;
    const size_t N1 = (size_t)NB * CIN * HWPIX * sizeof(float);   // 51,380,224 B per plane-set
    unsigned* mbuf = (unsigned*)ws;                               // 48 uints @ 0
    float* ux  = (float*)(ws + 256);
    float* uy  = (float*)(ws + 256 + 1 * N1);
    float* uxx = (float*)(ws + 256 + 2 * N1);
    float* uxy = (float*)(ws + 256 + 3 * N1);
    float* uyy = (float*)(ws + 256 + 4 * N1);
    _Float16* w1h = (_Float16*)(ws + 256 + 5 * N1);
    _Float16* w2h = (_Float16*)(ws + 256 + 5 * N1 + (size_t)KTOT * COUT * 2);

    // 0) zero max-abs accumulators
    invar_init_m<<<1, 64, 0, stream>>>(mbuf);

    // 1) depthwise derivative stencils from x
    const int convBlocks = NB * CIN * 49;     // 50176
    invar_conv7<<<convBlocks, 256, 0, stream>>>(x, ux, uy, uxx, uyy, mbuf, 0);

    // 2) uxy = kdx * uy (sequential, matches boundary behavior of reference)
    invar_conv7<<<convBlocks, 256, 0, stream>>>(uy, uxy, uxy, uxy, uxy, mbuf, 1);

    // 3) weights -> f16 (native [out][k] layout)
    invar_pack_w<<<(COUT * KTOT + 255) / 256, 256, 0, stream>>>(w1, w2, w1h, w2h);

    // 4) fused normalize + GEMM1 + ReLU + GEMM2
    const int gemmBlocks = NB * (HWPIX / 64);  // 16*196 = 3136
    invar_gemm<<<gemmBlocks, 256, 0, stream>>>(x, ux, uy, uxx, uxy, uyy,
                                               w1h, b1, w2h, b2, mbuf, out);
}
